// Field2point_18811956756749
// MI455X (gfx1250) — matrix operand order repaired and checked by reading, not compiled, not run on hardware
//
#include <hip/hip_runtime.h>
#include <math.h>

typedef __attribute__((ext_vector_type(16))) _Float16 v16h;
typedef __attribute__((ext_vector_type(8)))  _Float16 v8h;
typedef __attribute__((ext_vector_type(8)))  float    v8f;
typedef __attribute__((ext_vector_type(4)))  unsigned int u32x4;
typedef __attribute__((ext_vector_type(8)))  int          i32x8;
typedef __attribute__((ext_vector_type(4)))  int          i32x4;

#define B_DIM 16
#define L_DIM 8192
#define K_DIM 128
#define ROWP  136          // padded LDS row stride in halfs (272B = 68 dwords -> conflict-free b128 A loads)
#define PTS_PER_BLOCK 128  // 8 waves * 16 points
#define CHUNK_ROWS 32      // TDM staging chunk (32 rows x 128 f32 = 16 KB)

#define PI_F 3.14159265358979323846f

// Dirichlet kernel real part:
//   D_re(x, n) = (-1)^n * sin(64x) * cos(t)/sin(t),  t = x/2 - pi*n/128
// limit at sin(t)->0 is 128 (both singular points t=0 and t=-pi).
__device__ __forceinline__ float dirichlet_re(float sh, float ch, float s64,
                                              int n, const float* ct, const float* st) {
    float cn = ct[n], sn = st[n];
    float si = sh * cn - ch * sn;          // sin(t)
    float co = ch * cn + sh * sn;          // cos(t)
    float r  = s64 * co * __builtin_amdgcn_rcpf(si);
    r = (n & 1) ? -r : r;
    if (fabsf(si) < 1e-6f) r = 128.0f;     // exact limit at grid nodes
    return r;
}

__global__ __launch_bounds__(256)
void field2point_kernel(const float* __restrict__ x0,
                        const float* __restrict__ y0,
                        const float* __restrict__ psi,
                        float* __restrict__ out) {
    __shared__ _Float16 psiH[K_DIM * ROWP];        // 34,816 B: psi_b as f16, padded rows
    __shared__ float    stage[CHUNK_ROWS * K_DIM]; // 16 KB TDM staging buffer
    __shared__ float    ctab[K_DIM];               // cos(pi*n/128)
    __shared__ float    stab[K_DIM];               // sin(pi*n/128)
    __shared__ float    red[256];

    const int t    = threadIdx.x;
    const int b    = blockIdx.y;
    const int tile = blockIdx.x;

    // ---- tables ----
    if (t < K_DIM) {
        float ang = PI_F * (float)t * (1.0f / 128.0f);
        ctab[t] = cosf(ang);
        stab[t] = sinf(ang);
    }

    // ---- psi -> f16 LDS (via TDM), fused checkerboard reduction Q_b = sum (-1)^(m+n) psi ----
    const float* pb = psi + (size_t)b * K_DIM * K_DIM;
    float q = 0.0f;

#if defined(__gfx1250__) && __has_builtin(__builtin_amdgcn_tensor_load_to_lds)
    {
        const unsigned lds_off = (unsigned)(uintptr_t)(void*)stage;
        for (int c = 0; c < 4; ++c) {
            if (t < 32) {  // wave 0 drives the Tensor Data Mover (EXEC ignored, one issue per wave)
                unsigned long long ga =
                    (unsigned long long)(uintptr_t)(pb + c * CHUNK_ROWS * K_DIM);
                u32x4 g0; i32x8 g1; i32x4 gz4; i32x8 gz8;
                // D# group 0: count=1 | lds_addr | global_addr | type=2 ("image")
                g0[0] = 1u;
                g0[1] = lds_off;
                g0[2] = (unsigned)ga;
                g0[3] = (unsigned)((ga >> 32) & 0x1FFFFFFull) | (2u << 30);
                // D# group 1: 2-D tensor 128 x 32 of 4-byte elements, tile = whole chunk
                g1[0] = 0x20000;               // workgroup_mask=0, data_size=2 (4B)
                g1[1] = (K_DIM << 16);         // tensor_dim0 = 128 (lo16 in bits[63:48])
                g1[2] = (CHUNK_ROWS << 16);    // tensor_dim0 hi=0 | tensor_dim1 = 32
                g1[3] = (K_DIM << 16);         // tensor_dim1 hi=0 | tile_dim0 = 128
                g1[4] = CHUNK_ROWS;            // tile_dim1 = 32, tile_dim2 = 0
                g1[5] = K_DIM;                 // tensor_dim0_stride = 128
                g1[6] = 0;
                g1[7] = 0;
                gz4[0] = 0; gz4[1] = 0; gz4[2] = 0; gz4[3] = 0;
                #pragma unroll
                for (int j = 0; j < 8; ++j) gz8[j] = 0;
                // 6-arg form (clang-23 / therock headers): (g0, g1, g2, g3, g4, cpol)
                __builtin_amdgcn_tensor_load_to_lds(g0, g1, gz4, gz4, gz8, 0);
                __builtin_amdgcn_s_wait_tensorcnt(0);
            }
            __syncthreads();
            for (int i = t; i < CHUNK_ROWS * K_DIM; i += 256) {
                int m = c * CHUNK_ROWS + (i >> 7), n = i & 127;
                float v = stage[i];
                psiH[m * ROWP + n] = (_Float16)v;
                q = ((m + n) & 1) ? (q - v) : (q + v);
            }
            __syncthreads();
        }
    }
#else
    for (int i = t; i < K_DIM * K_DIM; i += 256) {
        int m = i >> 7, n = i & 127;
        float v = pb[i];
        psiH[m * ROWP + n] = (_Float16)v;
        q = ((m + n) & 1) ? (q - v) : (q + v);
    }
#endif

    red[t] = q;
    __syncthreads();
    #pragma unroll
    for (int s = 128; s > 0; s >>= 1) {
        if (t < s) red[t] += red[t + s];
        __syncthreads();
    }
    const float Q = red[0];

    // ---- per-wave point tile ----
    const int wave = t >> 5;
    const int lane = t & 31;
    const int lpt  = lane & 15;     // point within tile (N index)
    const int hi   = lane >> 4;     // which lane-half
    const int l0   = tile * PTS_PER_BLOCK + wave * 16;

    const float x = x0[(size_t)b * L_DIM + l0 + lpt];
    float sxh, cxh;
    sincosf(x * 0.5f, &sxh, &cxh);
    const float s64x = sinf(64.0f * x);

    // ---- accumulators U[m, l] for 8 M-tiles ----
    v8f acc[8];
    #pragma unroll
    for (int mt = 0; mt < 8; ++mt)
        #pragma unroll
        for (int r = 0; r < 8; ++r)
            acc[mt][r] = 0.0f;

    const int koffA = hi ? 16 : 0;   // A 16x32 f16 layout: lane-half K offset
    const int koffB = hi ? 16 : 0;   // B 32x16 f16 layout: lane-half K offset

    #pragma unroll
    for (int kt = 0; kt < 4; ++kt) {
        const int n0 = kt * 32;

        // B fragment: Bx[k = n0+koffB+e, N = lpt] = D_re(x_l, n)
        v16h bf;
        #pragma unroll
        for (int e = 0; e < 16; ++e)
            bf[e] = (_Float16)dirichlet_re(sxh, cxh, s64x, n0 + koffB + e, ctab, stab);

        #pragma unroll
        for (int mt = 0; mt < 8; ++mt) {
            const int mrow = mt * 16 + lpt;   // A row = M index (lanes 0-15 / 16-31 mirror)
            const v8h alo = *(const v8h*)&psiH[mrow * ROWP + n0 + koffA];
            const v8h ahi = *(const v8h*)&psiH[mrow * ROWP + n0 + 8 + koffA];
            v16h af;
            #pragma unroll
            for (int e = 0; e < 8; ++e) { af[e] = alo[e]; af[8 + e] = ahi[e]; }

            acc[mt] = __builtin_amdgcn_wmma_f32_16x16x32_f16(
                false, af, false, bf, (short)0, acc[mt], false, false);
        }
    }

    // ---- epilogue: out[l] = (sum_m Dy_re[l,m] * U[m,l] - sin64x*sin64y*Q) / K^2 ----
    const float y = y0[(size_t)b * L_DIM + l0 + lpt];
    float syh, cyh;
    sincosf(y * 0.5f, &syh, &cyh);
    const float s64y = sinf(64.0f * y);

    float P = 0.0f;
    #pragma unroll
    for (int mt = 0; mt < 8; ++mt) {
        #pragma unroll
        for (int r = 0; r < 8; ++r) {
            const int m = mt * 16 + (hi ? 8 : 0) + r;  // C/D layout: M = r + 8*hi
            P += dirichlet_re(syh, cyh, s64y, m, ctab, stab) * acc[mt][r];
        }
    }
    P += __shfl_xor(P, 16, 32);   // combine the two lane-halves (complementary M sets)

    if (lane < 16) {
        out[(size_t)b * L_DIM + l0 + lpt] =
            (P - s64x * s64y * Q) * (1.0f / 16384.0f);
    }
}

extern "C" void kernel_launch(void* const* d_in, const int* in_sizes, int n_in,
                              void* d_out, int out_size, void* d_ws, size_t ws_size,
                              hipStream_t stream) {
    const float* x0  = (const float*)d_in[0];
    const float* y0  = (const float*)d_in[1];
    const float* psi = (const float*)d_in[2];
    float* out = (float*)d_out;

    dim3 grid(L_DIM / PTS_PER_BLOCK, B_DIM);  // 64 x 16 blocks
    field2point_kernel<<<grid, 256, 0, stream>>>(x0, y0, psi, out);
}
